// HetGDC_12060268167480
// MI455X (gfx1250) — matrix operand order repaired and checked by reading, not compiled, not run on hardware
//
#include <hip/hip_runtime.h>
#include <hip/hip_bf16.h>
#include <math.h>

typedef __attribute__((ext_vector_type(16))) __bf16 v16bf;
typedef __attribute__((ext_vector_type(8)))  float  v8f;

#define N_NODES   100000
#define E_EDGES   1600000
#define DIM_IN    256
#define DIM_OUT   128
#define NTYPES    4
#define KSTEPS    10
#define ALPHA     0.1f
#define TEMP      2.0f
#define EPSN      1e-12f

static __device__ __forceinline__ __bf16 f2bf(float f) {
    union { float f; unsigned u; } v; v.f = f;
    unsigned r = (v.u + 0x7FFFu + ((v.u >> 16) & 1u)) >> 16;
    union { unsigned short s; __bf16 b; } o; o.s = (unsigned short)r;
    return o.b;
}

// ---------------------------------------------------------------------------
// 0) W [256,128] fp32 -> Wt [128,256] bf16 (row n holds K contiguous)
// ---------------------------------------------------------------------------
__global__ void wt_transpose_kernel(const float* __restrict__ W, __bf16* __restrict__ Wt) {
    int id = blockIdx.x * blockDim.x + threadIdx.x;
    if (id >= DIM_IN * DIM_OUT) return;
    int n = id >> 8;          // 0..127 output column
    int k = id & 255;         // 0..255 reduction index
    Wt[id] = f2bf(W[k * DIM_OUT + n]);
}

// ---------------------------------------------------------------------------
// 1) init: zero stats/accumulators, deg[i] = 1 (self loop)
// ---------------------------------------------------------------------------
__global__ void init_kernel(float* __restrict__ deg, float* __restrict__ stats, int nstats) {
    int i = blockIdx.x * blockDim.x + threadIdx.x;
    int stride = gridDim.x * blockDim.x;
    for (int j = i; j < N_NODES; j += stride) deg[j] = 1.0f;
    for (int j = i; j < nstats;  j += stride) stats[j] = 0.0f;
}

// ---------------------------------------------------------------------------
// 2) GEMM + bias + row L2-normalize:  H = l2norm(X @ W + b)
//    block = 256 threads (8 waves); wave owns a 16-row M tile; block = 128 rows.
//    Wt staged in LDS (64KB) via gfx1250 async global->LDS DMA, with a
//    32B-chunk XOR swizzle to spread DS banks on the B-tile reads.
// ---------------------------------------------------------------------------
__global__ __launch_bounds__(256)
void gemm_l2_kernel(const float* __restrict__ X, const __bf16* __restrict__ Wt,
                    const float* __restrict__ bias, float* __restrict__ H) {
    __shared__ __align__(32) __bf16 lds[DIM_OUT * DIM_IN];   // 65536 B

    const int tid = threadIdx.x;

    // async staged fill: element (n,k) -> n*256 + ((k/16 ^ (n&15))*16 + k%16)
    // each thread issues 16 async 16B copies; 8-element groups never straddle
    // a 16-element swizzle chunk, and both addresses stay 16B aligned.
    {
        const unsigned lds_base = (unsigned)(unsigned long long)(const void*)&lds[0];
        #pragma unroll
        for (int it = 0; it < 16; ++it) {
            int e  = (it * 256 + tid) * 8;      // linear bf16 element index
            int n  = e >> 8;
            int k0 = e & 255;
            unsigned ldsb = (unsigned)((n << 8) + (((k0 >> 4) ^ (n & 15)) << 4) + (k0 & 15)) * 2u;
            unsigned lds_addr = lds_base + ldsb;
            unsigned long long ga = (unsigned long long)(Wt + e);
            asm volatile("global_load_async_to_lds_b128 %0, %1, off"
                         :: "v"(lds_addr), "v"(ga) : "memory");
        }
        asm volatile("s_wait_asynccnt 0" ::: "memory");
    }
    __syncthreads();

    const int wave = tid >> 5;
    const int lane = tid & 31;
    const int half = lane >> 4;                 // 0/1
    const int ln   = lane & 15;                 // 0..15
    const int mbase = blockIdx.x * 128 + wave * 16;
    const int arow  = mbase + ln;
    const int aclmp = arow < N_NODES ? arow : N_NODES - 1;
    const float* xrow = X + (long)aclmp * DIM_IN;

    v8f acc[8];
    const v8f vz = {};
    #pragma unroll
    for (int i = 0; i < 8; ++i) acc[i] = vz;

    #pragma unroll
    for (int kt = 0; kt < 8; ++kt) {
        // A tile: lane(ln,half) holds row ln, K = kt*32 + 16h + e (contiguous)
        const int k0 = kt * 32 + half * 16;
        float4 xa = *(const float4*)(xrow + k0);
        float4 xb = *(const float4*)(xrow + k0 + 4);
        float4 xc = *(const float4*)(xrow + k0 + 8);
        float4 xd = *(const float4*)(xrow + k0 + 12);
        v16bf a;
        a[0]=f2bf(xa.x); a[1]=f2bf(xa.y); a[2]=f2bf(xa.z); a[3]=f2bf(xa.w);
        a[4]=f2bf(xb.x); a[5]=f2bf(xb.y); a[6]=f2bf(xb.z); a[7]=f2bf(xb.w);
        a[8]=f2bf(xc.x); a[9]=f2bf(xc.y); a[10]=f2bf(xc.z); a[11]=f2bf(xc.w);
        a[12]=f2bf(xd.x); a[13]=f2bf(xd.y); a[14]=f2bf(xd.z); a[15]=f2bf(xd.w);

        const int c0 = 2 * kt + half;           // 16-elem chunk index of kstart
        #pragma unroll
        for (int nt = 0; nt < 8; ++nt) {
            // B tile: lane(ln,half) = column nt*16+ln, K = kt*32 + 16h + e (contiguous)
            int n = nt * 16 + ln;
            const v16bf* bp = (const v16bf*)&lds[(n << 8) + ((c0 ^ (n & 15)) << 4)];
            acc[nt] = __builtin_amdgcn_wmma_f32_16x16x32_bf16(
                false, a, false, *bp, (short)0, acc[nt], false, false);
        }
    }

    // bias (depends on column only)
    #pragma unroll
    for (int nt = 0; nt < 8; ++nt) {
        float bn = bias[nt * 16 + ln];
        #pragma unroll
        for (int r = 0; r < 8; ++r) acc[nt][r] += bn;
    }

    // per-row L2 norm: row (r + 8*half) lives on the 16 lanes of this half
    float inv[8];
    #pragma unroll
    for (int r = 0; r < 8; ++r) {
        float p = 0.f;
        #pragma unroll
        for (int nt = 0; nt < 8; ++nt) p += acc[nt][r] * acc[nt][r];
        p += __shfl_xor(p, 1, 32);
        p += __shfl_xor(p, 2, 32);
        p += __shfl_xor(p, 4, 32);
        p += __shfl_xor(p, 8, 32);
        inv[r] = 1.0f / fmaxf(sqrtf(p), EPSN);
    }
    #pragma unroll
    for (int nt = 0; nt < 8; ++nt)
        #pragma unroll
        for (int r = 0; r < 8; ++r) acc[nt][r] *= inv[r];

    #pragma unroll
    for (int r = 0; r < 8; ++r) {
        int m = mbase + r + half * 8;
        if (m < N_NODES) {
            float* hrow = H + (long)m * DIM_OUT + ln;
            #pragma unroll
            for (int nt = 0; nt < 8; ++nt) hrow[nt * 16] = acc[nt][r];
        }
    }
}

// ---------------------------------------------------------------------------
// 3) per-type sums: g_s[t][d], g_ss[t][d], g_cnt[t]
// ---------------------------------------------------------------------------
__global__ void stats_kernel(const float* __restrict__ H, const int* __restrict__ ntype,
                             float* __restrict__ g_s, float* __restrict__ g_ss,
                             float* __restrict__ g_cnt) {
    int d = threadIdx.x;                               // 0..127
    float s0=0,s1=0,s2=0,s3=0, q0=0,q1=0,q2=0,q3=0;
    float c0=0,c1=0,c2=0,c3=0;
    for (int i = blockIdx.x; i < N_NODES; i += gridDim.x) {
        int t = ntype[i];
        float v = H[(long)i * DIM_OUT + d];
        float v2 = v * v;
        s0 += (t==0)?v:0.f;  q0 += (t==0)?v2:0.f;
        s1 += (t==1)?v:0.f;  q1 += (t==1)?v2:0.f;
        s2 += (t==2)?v:0.f;  q2 += (t==2)?v2:0.f;
        s3 += (t==3)?v:0.f;  q3 += (t==3)?v2:0.f;
        if (d == 0) { c0+=(t==0); c1+=(t==1); c2+=(t==2); c3+=(t==3); }
    }
    atomicAdd(&g_s[0*DIM_OUT+d], s0); atomicAdd(&g_ss[0*DIM_OUT+d], q0);
    atomicAdd(&g_s[1*DIM_OUT+d], s1); atomicAdd(&g_ss[1*DIM_OUT+d], q1);
    atomicAdd(&g_s[2*DIM_OUT+d], s2); atomicAdd(&g_ss[2*DIM_OUT+d], q2);
    atomicAdd(&g_s[3*DIM_OUT+d], s3); atomicAdd(&g_ss[3*DIM_OUT+d], q3);
    if (d == 0) {
        atomicAdd(&g_cnt[0], c0); atomicAdd(&g_cnt[1], c1);
        atomicAdd(&g_cnt[2], c2); atomicAdd(&g_cnt[3], c3);
    }
}

// 4) mean/std/istd from sums (unbiased variance like torch.std)
__global__ void finalize_stats_kernel(const float* __restrict__ g_s, const float* __restrict__ g_ss,
                                      const float* __restrict__ g_cnt, float* __restrict__ mean,
                                      float* __restrict__ stdv, float* __restrict__ istd) {
    int i = blockIdx.x * blockDim.x + threadIdx.x;
    if (i >= NTYPES * DIM_OUT) return;
    float c = g_cnt[i >> 7];
    float m = g_s[i] / c;
    float var = (g_ss[i] - c * m * m) / (c - 1.0f);
    float sd = sqrtf(fmaxf(var, 0.0f));
    mean[i] = m; stdv[i] = sd; istd[i] = 1.0f / sd;
}

// 5) tilde_H = (H - mean[type]) / std[type]
__global__ void tilde_kernel(const float* __restrict__ H, const int* __restrict__ ntype,
                             const float* __restrict__ mean, const float* __restrict__ istd,
                             float* __restrict__ tilde) {
    long i = (long)blockIdx.x * blockDim.x + threadIdx.x;
    long stride = (long)gridDim.x * blockDim.x;
    for (; i < (long)N_NODES * DIM_OUT; i += stride) {
        int row = (int)(i >> 7), d = (int)(i & 127);
        int t = ntype[row];
        tilde[i] = (H[i] - mean[t * DIM_OUT + d]) * istd[t * DIM_OUT + d];
    }
}

// 6) degree accumulation over dst (self-loop pre-seeded by init)
__global__ void degree_kernel(const int* __restrict__ dst, float* __restrict__ deg) {
    int i = blockIdx.x * blockDim.x + threadIdx.x;
    int stride = gridDim.x * blockDim.x;
    for (int e = i; e < E_EDGES; e += stride) atomicAdd(&deg[dst[e]], 1.0f);
}

// 7) deg -> d^{-1/2} in place
__global__ void dinv_kernel(float* __restrict__ deg) {
    int i = blockIdx.x * blockDim.x + threadIdx.x;
    int stride = gridDim.x * blockDim.x;
    for (int j = i; j < N_NODES; j += stride) deg[j] = rsqrtf(deg[j]);
}

// 8) edge weights norm[e] = dinv[src]*dinv[dst]
__global__ void wnorm_kernel(const int* __restrict__ src, const int* __restrict__ dst,
                             const float* __restrict__ dinv, float* __restrict__ wnorm) {
    int i = blockIdx.x * blockDim.x + threadIdx.x;
    int stride = gridDim.x * blockDim.x;
    for (int e = i; e < E_EDGES; e += stride) wnorm[e] = dinv[src[e]] * dinv[dst[e]];
}

// 9a) per-step dense init: h_next = alpha*tilde + (1-alpha)*dinv^2*h_cur (self loop)
__global__ void appnp_init_kernel(const float* __restrict__ h_cur, const float* __restrict__ tilde,
                                  const float* __restrict__ dinv, float* __restrict__ h_next) {
    long i = (long)blockIdx.x * blockDim.x + threadIdx.x;
    long stride = (long)gridDim.x * blockDim.x;
    for (; i < (long)N_NODES * DIM_OUT; i += stride) {
        float di = dinv[i >> 7];
        h_next[i] = ALPHA * tilde[i] + (1.0f - ALPHA) * di * di * h_cur[i];
    }
}

// 9b) per-step edge scatter: one wave per edge, float4 gather + 4 atomic adds
__global__ void appnp_scatter_kernel(const float* __restrict__ h_cur, float* __restrict__ h_next,
                                     const int* __restrict__ src, const int* __restrict__ dst,
                                     const float* __restrict__ wnorm) {
    int wid = (blockIdx.x * blockDim.x + threadIdx.x) >> 5;
    int lane = threadIdx.x & 31;
    int nw = (gridDim.x * blockDim.x) >> 5;
    for (int e = wid; e < E_EDGES; e += nw) {
        if (lane == 0 && e + nw < E_EDGES) {
            __builtin_prefetch(&src[e + nw], 0, 1);
            __builtin_prefetch(&dst[e + nw], 0, 1);
            __builtin_prefetch(&wnorm[e + nw], 0, 1);
        }
        int s = src[e], t = dst[e];
        float w = (1.0f - ALPHA) * wnorm[e];
        const float4 hv = *(const float4*)(h_cur + (long)s * DIM_OUT + lane * 4);
        float* out = h_next + (long)t * DIM_OUT + lane * 4;
        atomicAdd(out + 0, w * hv.x);
        atomicAdd(out + 1, w * hv.y);
        atomicAdd(out + 2, w * hv.z);
        atomicAdd(out + 3, w * hv.w);
    }
}

// 10) Z = tilde_Z * std[type] + mean[type]  -> d_out
__global__ void renorm_kernel(const float* __restrict__ hz, const int* __restrict__ ntype,
                              const float* __restrict__ mean, const float* __restrict__ stdv,
                              float* __restrict__ Z) {
    long i = (long)blockIdx.x * blockDim.x + threadIdx.x;
    long stride = (long)gridDim.x * blockDim.x;
    for (; i < (long)N_NODES * DIM_OUT; i += stride) {
        int row = (int)(i >> 7), d = (int)(i & 127);
        int t = ntype[row];
        Z[i] = hz[i] * stdv[t * DIM_OUT + d] + mean[t * DIM_OUT + d];
    }
}

// 11) z = l2norm(tilde_H): one wave per row
__global__ void znorm_kernel(const float* __restrict__ tilde, float* __restrict__ z) {
    int wid = (blockIdx.x * blockDim.x + threadIdx.x) >> 5;
    int lane = threadIdx.x & 31;
    int nw = (gridDim.x * blockDim.x) >> 5;
    for (int row = wid; row < N_NODES; row += nw) {
        const float4 v = *(const float4*)(tilde + (long)row * DIM_OUT + lane * 4);
        float p = v.x * v.x + v.y * v.y + v.z * v.z + v.w * v.w;
        p += __shfl_xor(p, 1, 32);  p += __shfl_xor(p, 2, 32);
        p += __shfl_xor(p, 4, 32);  p += __shfl_xor(p, 8, 32);
        p += __shfl_xor(p, 16, 32);
        float inv = 1.0f / fmaxf(sqrtf(p), EPSN);
        float4 o = { v.x * inv, v.y * inv, v.z * inv, v.w * inv };
        *(float4*)(z + (long)row * DIM_OUT + lane * 4) = o;
    }
}

// 12) sum_e exp(dot(z[a],z[b]) / TEMP) into *accum; one wave per edge
__global__ void loss_pairs_kernel(const float* __restrict__ z, const int* __restrict__ ea,
                                  const int* __restrict__ eb, float* __restrict__ accum) {
    int wid = (blockIdx.x * blockDim.x + threadIdx.x) >> 5;
    int lane = threadIdx.x & 31;
    int nw = (gridDim.x * blockDim.x) >> 5;
    float local = 0.0f;
    for (int e = wid; e < E_EDGES; e += nw) {
        int a = ea[e], b = eb[e];
        const float4 va = *(const float4*)(z + (long)a * DIM_OUT + lane * 4);
        const float4 vb = *(const float4*)(z + (long)b * DIM_OUT + lane * 4);
        float p = va.x * vb.x + va.y * vb.y + va.z * vb.z + va.w * vb.w;
        p += __shfl_xor(p, 1, 32);  p += __shfl_xor(p, 2, 32);
        p += __shfl_xor(p, 4, 32);  p += __shfl_xor(p, 8, 32);
        p += __shfl_xor(p, 16, 32);
        if (lane == 0) local += __expf(p * (1.0f / TEMP));
    }
    if (lane == 0) atomicAdd(accum, local);
}

// 13) loss = -log(pos/(pos+neg))
__global__ void loss_final_kernel(const float* __restrict__ pairsum, float* __restrict__ out_loss) {
    if (threadIdx.x == 0 && blockIdx.x == 0) {
        float pos = pairsum[0], neg = pairsum[1];
        out_loss[0] = -logf(pos / (pos + neg));
    }
}

// ---------------------------------------------------------------------------
extern "C" void kernel_launch(void* const* d_in, const int* in_sizes, int n_in,
                              void* d_out, int out_size, void* d_ws, size_t ws_size,
                              hipStream_t stream) {
    const float* X     = (const float*)d_in[0];
    const float* W     = (const float*)d_in[1];
    const float* bias  = (const float*)d_in[2];
    const int*   ntype = (const int*)d_in[3];
    const int*   e_src = (const int*)d_in[4];             // edge_index[0]
    const int*   e_dst = e_src + E_EDGES;                 // edge_index[1]
    const int*   n_src = (const int*)d_in[5];             // neg_edge[0]
    const int*   n_dst = n_src + E_EDGES;

    float* Zout = (float*)d_out;                          // N*128 floats
    float* Lout = Zout + (long)N_NODES * DIM_OUT;         // + loss scalar

    // workspace layout (256B aligned slabs)
    char* ws = (char*)d_ws;
    size_t o = 0;
    auto alloc = [&](size_t bytes) { char* p = ws + o; o = (o + bytes + 255) & ~(size_t)255; return p; };
    const size_t mat = (size_t)N_NODES * DIM_OUT * sizeof(float);
    float* H     = (float*)alloc(mat);                    // reused as z later
    float* tilde = (float*)alloc(mat);
    float* h0    = (float*)alloc(mat);
    float* h1    = (float*)alloc(mat);
    float* deg   = (float*)alloc((size_t)N_NODES * sizeof(float));   // becomes dinv
    float* wnorm = (float*)alloc((size_t)E_EDGES * sizeof(float));
    float* g_s   = (float*)alloc(NTYPES * DIM_OUT * sizeof(float));
    float* g_ss  = (float*)alloc(NTYPES * DIM_OUT * sizeof(float));
    float* g_cnt = (float*)alloc(NTYPES * sizeof(float));
    float* mean  = (float*)alloc(NTYPES * DIM_OUT * sizeof(float));
    float* stdv  = (float*)alloc(NTYPES * DIM_OUT * sizeof(float));
    float* istd  = (float*)alloc(NTYPES * DIM_OUT * sizeof(float));
    float* psum  = (float*)alloc(2 * sizeof(float));
    __bf16* Wt   = (__bf16*)alloc((size_t)DIM_IN * DIM_OUT * sizeof(__bf16));
    (void)ws_size; (void)n_in; (void)in_sizes; (void)out_size;

    // stats block is contiguous: g_s..psum -> zero all of it + deg=1
    int nstats = (int)(((char*)(psum + 2)) - (char*)g_s) / (int)sizeof(float);

    wt_transpose_kernel<<<(DIM_IN * DIM_OUT + 255) / 256, 256, 0, stream>>>(W, Wt);
    init_kernel<<<512, 256, 0, stream>>>(deg, g_s, nstats);

    gemm_l2_kernel<<<(N_NODES + 127) / 128, 256, 0, stream>>>(X, Wt, bias, H);

    stats_kernel<<<256, 128, 0, stream>>>(H, ntype, g_s, g_ss, g_cnt);
    finalize_stats_kernel<<<2, 256, 0, stream>>>(g_s, g_ss, g_cnt, mean, stdv, istd);
    tilde_kernel<<<1024, 256, 0, stream>>>(H, ntype, mean, istd, tilde);

    degree_kernel<<<1024, 256, 0, stream>>>(e_dst, deg);
    dinv_kernel<<<256, 256, 0, stream>>>(deg);
    wnorm_kernel<<<1024, 256, 0, stream>>>(e_src, e_dst, deg, wnorm);

    const float* cur = tilde;
    for (int step = 0; step < KSTEPS; ++step) {
        float* nxt = (step & 1) ? h1 : h0;     // step0->h0, step1->h1, ...
        appnp_init_kernel<<<1024, 256, 0, stream>>>(cur, tilde, deg, nxt);
        appnp_scatter_kernel<<<2048, 256, 0, stream>>>(cur, nxt, e_src, e_dst, wnorm);
        cur = nxt;
    }

    renorm_kernel<<<1024, 256, 0, stream>>>(cur, ntype, mean, stdv, Zout);

    float* z = H;  // H no longer needed
    znorm_kernel<<<2048, 256, 0, stream>>>(tilde, z);
    loss_pairs_kernel<<<2048, 256, 0, stream>>>(z, e_src, e_dst, psum + 0);
    loss_pairs_kernel<<<2048, 256, 0, stream>>>(z, n_src, n_dst, psum + 1);
    loss_final_kernel<<<1, 64, 0, stream>>>(psum, Lout);
}
